// DendriteLayer_19834158973323
// MI455X (gfx1250) — compile-verified, matrix operand order checked by reading, not run on hardware
//
#include <hip/hip_runtime.h>
#include <hip/hip_bf16.h>

// ---- problem constants (match the reference) ----
#define IN_DIM   1024
#define OUT_DIM  2048
#define DPN      16
#define BATCH    256

// ---- tiling ----
#define UNITS    4                 // output units per block (N tile = 64)
#define ROWSW    (UNITS * DPN)     // 64 weight rows staged per block
#define KC       128               // K-chunk staged in LDS per iteration
#define LDSP     (KC + 8)          // padded K stride (halves) -> conflict-free ds reads
#define THREADS  256               // 8 waves (wave32)

// sched_group_barrier instruction-class masks
#define SGB_WMMA     0x008
#define SGB_VMEM_RD  0x020
#define SGB_DS_RD    0x100

typedef __attribute__((ext_vector_type(16))) __bf16 v16bf;
typedef __attribute__((ext_vector_type(8)))  __bf16 v8bf;
typedef __attribute__((ext_vector_type(8)))  float  v8f;

__device__ __forceinline__ v8bf cvt8(float4 a, float4 b) {
    v8bf r;
    r[0] = (__bf16)a.x; r[1] = (__bf16)a.y; r[2] = (__bf16)a.z; r[3] = (__bf16)a.w;
    r[4] = (__bf16)b.x; r[5] = (__bf16)b.y; r[6] = (__bf16)b.z; r[7] = (__bf16)b.w;
    return r;
}

__device__ __forceinline__ v16bf join16(v8bf lo, v8bf hi) {
    return __builtin_shufflevector(lo, hi,
        0, 1, 2, 3, 4, 5, 6, 7, 8, 9, 10, 11, 12, 13, 14, 15);
}

// ---- pre-pass: x f32 -> bf16 once (removes per-block conversion VALU) ----
__global__ __launch_bounds__(THREADS)
void cvt_x_kernel(const float* __restrict__ x, __bf16* __restrict__ xb) {
    const int i = (blockIdx.x * THREADS + threadIdx.x) * 8;
    float4 a = *(const float4*)(x + i);
    float4 b = *(const float4*)(x + i + 4);
    *(v8bf*)(xb + i) = cvt8(a, b);
}

// One block per group of 4 output units. Block = 8 waves; wave w computes batch
// rows [32w, 32w+32) against the group's 64 dendrites (4 WMMA N-tiles sharing
// each A fragment), then does the per-unit k=1 winner-take-all in-wave.
template <bool XBF>
__global__ __launch_bounds__(THREADS)
void dendrite_main(const void* __restrict__ xsrc,
                   const float* __restrict__ w_in,
                   const float* __restrict__ b_in,
                   const float* __restrict__ w_out,
                   const float* __restrict__ b_out,
                   float* __restrict__ out) {
    __shared__ __align__(16) __bf16 wlds[2][ROWSW * LDSP];  // ping-pong bf16 weights

    const int tid  = threadIdx.x;
    const int lane = tid & 31;
    const int wave = tid >> 5;          // 0..7
    const int g    = blockIdx.x;        // unit group (units g*4 .. g*4+3)
    const int d0   = g * ROWSW;         // first dendrite row of this block

    // ---- weight staging map: 512 strips of 16 floats; 2 strips per thread ----
    const int row0 = tid >> 3;                   // strip j=0: rows 0..31
    const int ko0  = (tid & 7) * 16;
    const int row1 = row0 + 32;                  // strip j=1: rows 32..63
    const float* wsrc0 = w_in + (size_t)(d0 + row0) * IN_DIM + ko0;
    const float* wsrc1 = w_in + (size_t)(d0 + row1) * IN_DIM + ko0;

    // ---- fragment addressing (ISA 16-bit A/B layout for 16x16x32) ----
    const int fcol = lane & 15;                  // A: M row ; B: dendrite col
    const int koff = (lane < 16) ? 0 : 8;        // low/high half-wave K offset
    const int m0   = wave * 32;

    const float*  xf = (const float*)xsrc;
    const __bf16* xb = (const __bf16*)xsrc;
    const size_t roff0 = (size_t)(m0 + fcol) * IN_DIM;        // tile 0 row
    const size_t roff1 = roff0 + (size_t)16 * IN_DIM;         // tile 1 row

    v8f acc[UNITS][2];
    #pragma unroll
    for (int q = 0; q < UNITS; ++q) { acc[q][0] = (v8f){}; acc[q][1] = (v8f){}; }

    // prefetch first weight chunk into registers (2 strips x 4 float4)
    float4 pf[2][4];
    #pragma unroll
    for (int i = 0; i < 4; ++i) pf[0][i] = *(const float4*)(wsrc0 + i * 4);
    #pragma unroll
    for (int i = 0; i < 4; ++i) pf[1][i] = *(const float4*)(wsrc1 + i * 4);

    for (int kc = 0; kc < IN_DIM; kc += KC) {
        const int buf = (kc / KC) & 1;

        // stage prefetched chunk -> LDS as bf16
        __bf16* wd0 = &wlds[buf][row0 * LDSP + ko0];
        __bf16* wd1 = &wlds[buf][row1 * LDSP + ko0];
        *(v8bf*)(wd0)     = cvt8(pf[0][0], pf[0][1]);
        *(v8bf*)(wd0 + 8) = cvt8(pf[0][2], pf[0][3]);
        *(v8bf*)(wd1)     = cvt8(pf[1][0], pf[1][1]);
        *(v8bf*)(wd1 + 8) = cvt8(pf[1][2], pf[1][3]);

        // prefetch next chunk while this one is consumed
        if (kc + KC < IN_DIM) {
            #pragma unroll
            for (int i = 0; i < 4; ++i) pf[0][i] = *(const float4*)(wsrc0 + kc + KC + i * 4);
            #pragma unroll
            for (int i = 0; i < 4; ++i) pf[1][i] = *(const float4*)(wsrc1 + kc + KC + i * 4);
        }

        __syncthreads();   // staged tile visible; other buffer's readers already synced

        #pragma unroll
        for (int k = 0; k < KC; k += 32) {
            const int gk = kc + k;

            // A fragments first (global latency overlaps the B ds-loads below)
            v16bf af0, af1;
            if (XBF) {
                af0 = join16(*(const v8bf*)(xb + roff0 + gk + koff),
                             *(const v8bf*)(xb + roff0 + gk + 16 + koff));
                af1 = join16(*(const v8bf*)(xb + roff1 + gk + koff),
                             *(const v8bf*)(xb + roff1 + gk + 16 + koff));
            } else {
                const float* r0 = xf + roff0 + gk;
                const float* r1 = xf + roff1 + gk;
                af0 = join16(cvt8(*(const float4*)(r0 + koff),
                                  *(const float4*)(r0 + koff + 4)),
                             cvt8(*(const float4*)(r0 + 16 + koff),
                                  *(const float4*)(r0 + 16 + koff + 4)));
                af1 = join16(cvt8(*(const float4*)(r1 + koff),
                                  *(const float4*)(r1 + koff + 4)),
                             cvt8(*(const float4*)(r1 + 16 + koff),
                                  *(const float4*)(r1 + 16 + koff + 4)));
            }

            // all B fragments in distinct registers
            v16bf bf[UNITS];
            #pragma unroll
            for (int q = 0; q < UNITS; ++q) {
                const __bf16* wb = &wlds[buf][(q * DPN + fcol) * LDSP + k + koff];
                bf[q] = join16(*(const v8bf*)(wb), *(const v8bf*)(wb + 16));
            }

            #pragma unroll
            for (int q = 0; q < UNITS; ++q) {
                acc[q][0] = __builtin_amdgcn_wmma_f32_16x16x32_bf16(
                                false, af0, false, bf[q], (short)0, acc[q][0], false, false);
                acc[q][1] = __builtin_amdgcn_wmma_f32_16x16x32_bf16(
                                false, af1, false, bf[q], (short)0, acc[q][1], false, false);
            }

            if (XBF) {
                // Pin the per-k-step schedule: 4 global A reads, then all 8 B
                // ds-reads (one clause, distinct regs), then 8 back-to-back
                // WMMAs. The waitcnt pass can then use descending s_wait_dscnt
                // thresholds instead of a full dscnt=0 stall per WMMA pair.
                __builtin_amdgcn_sched_group_barrier(SGB_VMEM_RD, 4, 0);
                __builtin_amdgcn_sched_group_barrier(SGB_DS_RD,   8, 0);
                __builtin_amdgcn_sched_group_barrier(SGB_WMMA,    8, 0);
            }
        }
        // next iteration writes the other buffer; its __syncthreads orders us.
    }

    // ---- epilogue: h = c + b_in ; winner over 16 dendrites (lanes of each
    //      16-lane half) ; out = h* * w_out[u, argmax] + b_out
    const int rowbase = m0 + ((lane < 16) ? 0 : 8);

    #pragma unroll
    for (int q = 0; q < UNITS; ++q) {
        const int u = g * UNITS + q;
        const float binv = b_in[u * DPN + fcol];
        const float wov  = w_out[u * DPN + fcol];
        const float bov  = b_out[u];
        #pragma unroll
        for (int t = 0; t < 2; ++t) {
            #pragma unroll
            for (int r = 0; r < 8; ++r) {
                float h  = acc[q][t][r] + binv;
                int  idx = fcol;
                #pragma unroll
                for (int off = 8; off >= 1; off >>= 1) {
                    float oh = __shfl_xor(h, off, 32);
                    int   oi = __shfl_xor(idx, off, 32);
                    // strict-greater wins; tie keeps smallest index (jnp.argmax)
                    bool take = (oh > h) || ((oh == h) && (oi < idx));
                    if (take) { h = oh; idx = oi; }
                }
                float wsel = __shfl(wov, idx, 32);   // winner's w_out (same per half)
                if (fcol == 0) {
                    const int row = rowbase + t * 16 + r;
                    out[(size_t)row * OUT_DIM + u] = h * wsel + bov;
                }
            }
        }
    }
}

extern "C" void kernel_launch(void* const* d_in, const int* in_sizes, int n_in,
                              void* d_out, int out_size, void* d_ws, size_t ws_size,
                              hipStream_t stream) {
    (void)in_sizes; (void)n_in; (void)out_size;
    const float* x     = (const float*)d_in[0];   // [256, 1024]
    const float* w_in  = (const float*)d_in[1];   // [32768, 1024]
    const float* b_in  = (const float*)d_in[2];   // [32768]
    const float* w_out = (const float*)d_in[3];   // [2048, 16]
    const float* b_out = (const float*)d_in[4];   // [2048]
    float* out = (float*)d_out;                   // [256, 2048]

    const size_t xb_bytes = (size_t)BATCH * IN_DIM * sizeof(__bf16);
    if (ws_size >= xb_bytes) {
        __bf16* xb = (__bf16*)d_ws;
        cvt_x_kernel<<<(BATCH * IN_DIM) / (THREADS * 8), THREADS, 0, stream>>>(x, xb);
        dendrite_main<true><<<OUT_DIM / UNITS, THREADS, 0, stream>>>(
            xb, w_in, b_in, w_out, b_out, out);
    } else {
        dendrite_main<false><<<OUT_DIM / UNITS, THREADS, 0, stream>>>(
            x, w_in, b_in, w_out, b_out, out);
    }
}
